// SelfAttention_46196668235916
// MI455X (gfx1250) — matrix operand order repaired and checked by reading, not compiled, not run on hardware
//
#include <hip/hip_runtime.h>

#define NHEAD 4
#define LTOK  4096   // H*W
#define DHEAD 64
#define CCH   256

typedef __bf16 bf16;
typedef __attribute__((ext_vector_type(8)))  __bf16 v8bf;
typedef __attribute__((ext_vector_type(16))) __bf16 v16bf;
typedef __attribute__((ext_vector_type(8)))  float  v8f;

// Build a 16-element bf16 fragment from two aligned 16-byte chunks.
__device__ __forceinline__ v16bf frag2(const bf16* p0, const bf16* p1) {
  v8bf a = *(const v8bf*)p0;
  v8bf b = *(const v8bf*)p1;
  return __builtin_shufflevector(a, b, 0,1,2,3,4,5,6,7,8,9,10,11,12,13,14,15);
}

__device__ __forceinline__ v8f wmma_bf16(v16bf a, v16bf b, v8f c) {
  // D = A(16x32) * B(32x16) + C(16x16 f32)
  return __builtin_amdgcn_wmma_f32_16x16x32_bf16(false, a, false, b, (short)0, c,
                                                 false, false);
}

// Low 32 bits of a flat shared-memory pointer == LDS byte address (ISA 10.2).
__device__ __forceinline__ unsigned lds_addr(const void* p) {
  return (unsigned)(uintptr_t)p;
}

// CDNA5 async copy: global -> LDS, 16 bytes per lane, tracked by ASYNCcnt.
// INST_OFFSET is added to BOTH the LDS and global addresses (ISA 10.7/15.18.3),
// so one base pair covers consecutive 16-byte chunks via offset:16.
__device__ __forceinline__ void async_copy_b128(unsigned lds, const void* gsrc) {
  asm volatile("global_load_async_to_lds_b128 %0, %1, off"
               :: "v"(lds), "v"((unsigned long long)(uintptr_t)gsrc) : "memory");
}
__device__ __forceinline__ void async_copy_b128_off16(unsigned lds, const void* gsrc) {
  asm volatile("global_load_async_to_lds_b128 %0, %1, off offset:16"
               :: "v"(lds), "v"((unsigned long long)(uintptr_t)gsrc) : "memory");
}
__device__ __forceinline__ void wait_asynccnt0() {
  asm volatile("s_wait_asynccnt 0x0" ::: "memory");
}

// ---------------------------------------------------------------------------
// K1: GroupNorm statistics -> per (b, channel) scale s and shift t:
//     norm = x * s[b,c] + t[b,c]
// ---------------------------------------------------------------------------
__global__ void __launch_bounds__(256)
gn_stats_kernel(const float* __restrict__ x,
                const float* __restrict__ gamma,
                const float* __restrict__ beta,
                float* __restrict__ sArr, float* __restrict__ tArr) {
  int bg = blockIdx.x;            // 0..127  (B * 32 groups)
  int b = bg >> 5, g = bg & 31;
  const float* base = x + (size_t)(b * CCH + g * 8) * LTOK;
  float sum = 0.f, sq = 0.f;
  for (int i = threadIdx.x; i < 8 * LTOK; i += 256) {
    float v = base[i];
    sum += v; sq += v * v;
  }
  __shared__ float s1[256], s2[256];
  s1[threadIdx.x] = sum; s2[threadIdx.x] = sq;
  __syncthreads();
  for (int off = 128; off > 0; off >>= 1) {
    if ((int)threadIdx.x < off) {
      s1[threadIdx.x] += s1[threadIdx.x + off];
      s2[threadIdx.x] += s2[threadIdx.x + off];
    }
    __syncthreads();
  }
  if (threadIdx.x < 8) {
    float mean = s1[0] * (1.f / 32768.f);
    float var  = s2[0] * (1.f / 32768.f) - mean * mean;
    float rstd = rsqrtf(var + 1e-5f);
    int c = g * 8 + threadIdx.x;
    float s = rstd * gamma[c];
    sArr[b * CCH + c] = s;
    tArr[b * CCH + c] = beta[c] - mean * s;
  }
}

// ---------------------------------------------------------------------------
// K2: QKV projection. qkv[o,p] = sum_c wqkv[o,c] * norm[c,p] + bqkv[o]
// GEMM M=768 N=4096 K=256 per batch. Block tile 64x128, 8 waves (4x2),
// each wave: one 16-row strip x four 16-col tiles. Writes bf16:
//   Q token-major [b,h][L][64]  (pre-scaled by 1/sqrt(256) = 1/16)
//   K token-major [b,h][L][64]
//   V d-major     [b,h][64][L]
// ---------------------------------------------------------------------------
__global__ void __launch_bounds__(256)
qkv_kernel(const float* __restrict__ x,
           const float* __restrict__ wqkv,
           const float* __restrict__ bqkv,
           const float* __restrict__ sArr,
           const float* __restrict__ tArr,
           bf16* __restrict__ Qws, bf16* __restrict__ Kws, bf16* __restrict__ Vws) {
  __shared__ __align__(16) bf16 Alds[64][40];    // [m][k]
  __shared__ __align__(16) bf16 Blds[128][40];   // [n][k]  (k contiguous per column)
  int p0 = blockIdx.x * 128;
  int o0 = blockIdx.y * 64;
  int b  = blockIdx.z;
  int tid = threadIdx.x, lane = tid & 31, w = tid >> 5;
  int wm = w & 3, wn = w >> 2, n15 = lane & 15, hi = lane >> 4;

  v8f zero = {0.f,0.f,0.f,0.f,0.f,0.f,0.f,0.f};
  v8f acc[4] = {zero, zero, zero, zero};

  for (int kc = 0; kc < CCH; kc += 32) {
    { // stage A: weights f32 -> bf16
      int m = tid >> 2, kch = (tid & 3) * 8;
      const float* src = wqkv + (size_t)(o0 + m) * CCH + kc + kch;
      v8bf tmp;
#pragma unroll
      for (int i = 0; i < 8; i++) tmp[i] = (bf16)src[i];
      *(v8bf*)&Alds[m][kch] = tmp;
    }
    { // stage B: groupnorm fused, transposed to [n][k]
      int k = tid >> 3, nch = (tid & 7) * 16;
      int c = kc + k;
      float s = sArr[b * CCH + c], tt = tArr[b * CCH + c];
      const float* src = x + (size_t)(b * CCH + c) * LTOK + p0 + nch;
#pragma unroll
      for (int i = 0; i < 16; i++) Blds[nch + i][k] = (bf16)(src[i] * s + tt);
    }
    __syncthreads();
    v16bf afrag = frag2(&Alds[wm * 16 + n15][hi * 16],
                        &Alds[wm * 16 + n15][hi * 16 + 8]);
#pragma unroll
    for (int t = 0; t < 4; t++) {
      int col = wn * 64 + t * 16 + n15;
      v16bf bfr = frag2(&Blds[col][hi * 16], &Blds[col][hi * 16 + 8]);
      acc[t] = wmma_bf16(afrag, bfr, acc[t]);
    }
    __syncthreads();
  }
  // epilogue: route rows to Q/K/V with their layouts
#pragma unroll
  for (int t = 0; t < 4; t++) {
    int p = p0 + wn * 64 + t * 16 + n15;
#pragma unroll
    for (int r = 0; r < 8; r++) {
      int o = o0 + wm * 16 + r + hi * 8;
      float v = acc[t][r] + bqkv[o];
      int head = o / 192, rem = o % 192;
      size_t bh = (size_t)(b * NHEAD + head);
      if (rem < 64) {
        Qws[(bh * LTOK + p) * DHEAD + rem] = (bf16)(v * 0.0625f);
      } else if (rem < 128) {
        Kws[(bh * LTOK + p) * DHEAD + (rem - 64)] = (bf16)v;
      } else {
        Vws[(bh * DHEAD + (rem - 128)) * LTOK + p] = (bf16)v;
      }
    }
  }
}

// ---------------------------------------------------------------------------
// K3: flash attention per (b, head). Block = 128 queries (8 waves x 16),
// 64-key tiles double-buffered in LDS via async global->LDS copies (ASYNCcnt),
// online softmax, O accumulators in registers.
// Writes attention output bf16 channel-major [b][c][p] for the out-proj GEMM.
// ---------------------------------------------------------------------------
__global__ void __launch_bounds__(256)
attn_kernel(const bf16* __restrict__ Qws, const bf16* __restrict__ Kws,
            const bf16* __restrict__ Vws, bf16* __restrict__ Ofeat) {
  __shared__ __align__(16) bf16 Klds[2][64][72];    // [buf][key][d]
  __shared__ __align__(16) bf16 Vlds[2][64][72];    // [buf][d][key]
  __shared__ __align__(16) bf16 Plds[8][16][72];    // per-wave P tile [row][key]

  int head = blockIdx.y, b = blockIdx.z;
  size_t bh = (size_t)(b * NHEAD + head);
  int tid = threadIdx.x, lane = tid & 31, w = tid >> 5;
  int n15 = lane & 15, hi = lane >> 4;
  int i0 = blockIdx.x * 128 + w * 16;

  // per-thread staging coordinates (32 contiguous bytes per thread per tile)
  int kjj = tid >> 2, kdch = (tid & 3) * 16;   // K: [key][d]
  int vdd = tid >> 2, vjch = (tid & 3) * 16;   // V: [d][key]
  const bf16* Kgsrc = Kws + (bh * LTOK + kjj) * DHEAD + kdch;   // + jb*DHEAD
  const bf16* Vgsrc = Vws + (bh * DHEAD + vdd) * LTOK + vjch;   // + jb

  // Q fragments held in registers for the whole key loop (2 K-steps of 32)
  const bf16* Qbase = Qws + (bh * LTOK + i0) * DHEAD;
  v16bf qfrag[2];
#pragma unroll
  for (int ks = 0; ks < 2; ks++) {
    const bf16* p = Qbase + (size_t)n15 * DHEAD + ks * 32 + hi * 16;
    qfrag[ks] = frag2(p, p + 8);
  }

  v8f zero = {0.f,0.f,0.f,0.f,0.f,0.f,0.f,0.f};
  v8f acc[4] = {zero, zero, zero, zero};
  v8f mrow, lrow;
#pragma unroll
  for (int r = 0; r < 8; r++) { mrow[r] = -1e30f; lrow[r] = 0.f; }

  // prologue: async-stage tile jb=0 into buffer 0
  {
    unsigned lK = lds_addr(&Klds[0][kjj][kdch]);
    unsigned lV = lds_addr(&Vlds[0][vdd][vjch]);
    async_copy_b128(lK, Kgsrc);
    async_copy_b128_off16(lK, Kgsrc);
    async_copy_b128(lV, Vgsrc);
    async_copy_b128_off16(lV, Vgsrc);
  }
  wait_asynccnt0();
  __syncthreads();

  for (int jb = 0; jb < LTOK; jb += 64) {
    int cur = (jb >> 6) & 1;
    // issue async loads for the NEXT tile into the other buffer; they run in
    // the background while this tile's WMMAs + softmax execute.
    if (jb + 64 < LTOK) {
      int nxt = cur ^ 1;
      unsigned lK = lds_addr(&Klds[nxt][kjj][kdch]);
      unsigned lV = lds_addr(&Vlds[nxt][vdd][vjch]);
      async_copy_b128(lK, Kgsrc + (size_t)(jb + 64) * DHEAD);
      async_copy_b128_off16(lK, Kgsrc + (size_t)(jb + 64) * DHEAD);
      async_copy_b128(lV, Vgsrc + (jb + 64));
      async_copy_b128_off16(lV, Vgsrc + (jb + 64));
    }

    // S = Q * K^T : 4 column tiles x 2 K-steps
    v8f s[4];
#pragma unroll
    for (int t = 0; t < 4; t++) {
      v8f sv = zero;
#pragma unroll
      for (int ks = 0; ks < 2; ks++) {
        const bf16* kp = &Klds[cur][t * 16 + n15][ks * 32 + hi * 16];
        sv = wmma_bf16(qfrag[ks], frag2(kp, kp + 8), sv);
      }
      s[t] = sv;
    }

    // online softmax: row max / exp / row sum (rows live per 16-lane half)
    v8f mn, scale;
#pragma unroll
    for (int r = 0; r < 8; r++) {
      float v = fmaxf(fmaxf(s[0][r], s[1][r]), fmaxf(s[2][r], s[3][r]));
      v = fmaxf(v, __shfl_xor(v, 1, 32));
      v = fmaxf(v, __shfl_xor(v, 2, 32));
      v = fmaxf(v, __shfl_xor(v, 4, 32));
      v = fmaxf(v, __shfl_xor(v, 8, 32));
      mn[r] = fmaxf(mrow[r], v);
      scale[r] = __expf(mrow[r] - mn[r]);
    }
    mrow = mn;
#pragma unroll
    for (int r = 0; r < 8; r++) {
      float rsum = 0.f;
#pragma unroll
      for (int t = 0; t < 4; t++) {
        float p = __expf(s[t][r] - mn[r]);
        s[t][r] = p;
        rsum += p;
      }
      rsum += __shfl_xor(rsum, 1, 32);
      rsum += __shfl_xor(rsum, 2, 32);
      rsum += __shfl_xor(rsum, 4, 32);
      rsum += __shfl_xor(rsum, 8, 32);
      lrow[r] = lrow[r] * scale[r] + rsum;
#pragma unroll
      for (int t = 0; t < 4; t++) acc[t][r] *= scale[r];
    }

    // P (C-layout) -> LDS -> A-layout fragments (per-wave region, no barrier)
#pragma unroll
    for (int t = 0; t < 4; t++)
#pragma unroll
      for (int r = 0; r < 8; r++)
        Plds[w][r + hi * 8][t * 16 + n15] = (bf16)s[t][r];

    // O += P * V : 2 K-steps (keys) x 4 d-tiles
#pragma unroll
    for (int ks = 0; ks < 2; ks++) {
      const bf16* pp = &Plds[w][n15][ks * 32 + hi * 16];
      v16bf pfrag = frag2(pp, pp + 8);
#pragma unroll
      for (int t = 0; t < 4; t++) {
        const bf16* vp = &Vlds[cur][t * 16 + n15][ks * 32 + hi * 16];
        acc[t] = wmma_bf16(pfrag, frag2(vp, vp + 8), acc[t]);
      }
    }

    // wave's own async loads for the next tile must land before the barrier;
    // the barrier also guarantees no wave still reads the buffer we overwrite.
    wait_asynccnt0();
    __syncthreads();
  }

  // epilogue: divide by row sum, write channel-major bf16
#pragma unroll
  for (int t = 0; t < 4; t++) {
    int c = head * 64 + t * 16 + n15;
#pragma unroll
    for (int r = 0; r < 8; r++) {
      int i = i0 + r + hi * 8;
      Ofeat[(size_t)(b * CCH + c) * LTOK + i] = (bf16)(acc[t][r] / lrow[r]);
    }
  }
}

// ---------------------------------------------------------------------------
// K4: out projection + bias + residual + 1/sqrt(2).
// GEMM M=256 N=4096 K=256 per batch, same tiling as K2.
// ---------------------------------------------------------------------------
__global__ void __launch_bounds__(256)
outproj_kernel(const float* __restrict__ x,
               const float* __restrict__ wout,
               const float* __restrict__ bout,
               const bf16* __restrict__ Ofeat,
               float* __restrict__ out) {
  __shared__ __align__(16) bf16 Alds[64][40];
  __shared__ __align__(16) bf16 Blds[128][40];
  int p0 = blockIdx.x * 128, o0 = blockIdx.y * 64, b = blockIdx.z;
  int tid = threadIdx.x, lane = tid & 31, w = tid >> 5;
  int wm = w & 3, wn = w >> 2, n15 = lane & 15, hi = lane >> 4;

  v8f zero = {0.f,0.f,0.f,0.f,0.f,0.f,0.f,0.f};
  v8f acc[4] = {zero, zero, zero, zero};

  for (int kc = 0; kc < CCH; kc += 32) {
    {
      int m = tid >> 2, kch = (tid & 3) * 8;
      const float* src = wout + (size_t)(o0 + m) * CCH + kc + kch;
      v8bf tmp;
#pragma unroll
      for (int i = 0; i < 8; i++) tmp[i] = (bf16)src[i];
      *(v8bf*)&Alds[m][kch] = tmp;
    }
    {
      int k = tid >> 3, nch = (tid & 7) * 16;
      const bf16* src = Ofeat + (size_t)(b * CCH + kc + k) * LTOK + p0 + nch;
      v8bf t0 = *(const v8bf*)src;
      v8bf t1 = *(const v8bf*)(src + 8);
#pragma unroll
      for (int i = 0; i < 8; i++) Blds[nch + i][k] = t0[i];
#pragma unroll
      for (int i = 0; i < 8; i++) Blds[nch + 8 + i][k] = t1[i];
    }
    __syncthreads();
    v16bf afrag = frag2(&Alds[wm * 16 + n15][hi * 16],
                        &Alds[wm * 16 + n15][hi * 16 + 8]);
#pragma unroll
    for (int t = 0; t < 4; t++) {
      int col = wn * 64 + t * 16 + n15;
      v16bf bfr = frag2(&Blds[col][hi * 16], &Blds[col][hi * 16 + 8]);
      acc[t] = wmma_bf16(afrag, bfr, acc[t]);
    }
    __syncthreads();
  }
#pragma unroll
  for (int t = 0; t < 4; t++) {
    int p = p0 + wn * 64 + t * 16 + n15;
#pragma unroll
    for (int r = 0; r < 8; r++) {
      int o = o0 + wm * 16 + r + hi * 8;
      size_t idx = (size_t)(b * CCH + o) * LTOK + p;
      float v = acc[t][r] + bout[o];
      out[idx] = (v + x[idx]) * 0.70710678118654752f;
    }
  }
}

// ---------------------------------------------------------------------------
extern "C" void kernel_launch(void* const* d_in, const int* in_sizes, int n_in,
                              void* d_out, int out_size, void* d_ws, size_t ws_size,
                              hipStream_t stream) {
  (void)in_sizes; (void)n_in; (void)out_size; (void)ws_size;
  const float* x     = (const float*)d_in[0];
  const float* gamma = (const float*)d_in[1];
  const float* beta  = (const float*)d_in[2];
  const float* wqkv  = (const float*)d_in[3];
  const float* bqkv  = (const float*)d_in[4];
  const float* wout  = (const float*)d_in[5];
  const float* bout  = (const float*)d_in[6];
  float* out = (float*)d_out;

  char* ws = (char*)d_ws;
  const size_t QKV_BYTES = (size_t)4 * NHEAD * LTOK * DHEAD * sizeof(bf16); // 8 MiB each
  float* sArr = (float*)(ws);
  float* tArr = (float*)(ws + 4096);
  bf16* Qws   = (bf16*)(ws + 8192);
  bf16* Kws   = (bf16*)(ws + 8192 + QKV_BYTES);
  bf16* Vws   = (bf16*)(ws + 8192 + 2 * QKV_BYTES);
  bf16* Ofeat = (bf16*)(ws + 8192 + 3 * QKV_BYTES);

  gn_stats_kernel<<<128, 256, 0, stream>>>(x, gamma, beta, sArr, tArr);
  qkv_kernel<<<dim3(32, 12, 4), 256, 0, stream>>>(x, wqkv, bqkv, sArr, tArr,
                                                  Qws, Kws, Vws);
  attn_kernel<<<dim3(32, NHEAD, 4), 256, 0, stream>>>(Qws, Kws, Vws, Ofeat);
  outproj_kernel<<<dim3(32, 4, 4), 256, 0, stream>>>(x, wout, bout, Ofeat, out);
}
